// TransformerEncoder_53790170415155
// MI455X (gfx1250) — matrix operand order repaired and checked by reading, not compiled, not run on hardware
//
#include <hip/hip_runtime.h>
#include <hip/hip_bf16.h>
#include <math.h>

// ---- problem constants (match reference) ----
constexpr int LYR = 6;
constexpr int DIM = 1024;
constexpr int FF  = 4096;
constexpr int NH  = 16;
constexpr int KW  = 128;
constexpr int HDm = 64;     // DIM / NH
constexpr int Bb  = 8;
constexpr int Tt  = 512;
constexpr int Mrows = Bb * Tt;  // 4096

// ---- types ----
typedef __bf16 bf16_t;
typedef __attribute__((ext_vector_type(8)))  bf16_t v8bf;
typedef __attribute__((ext_vector_type(16))) bf16_t v16bf;
typedef __attribute__((ext_vector_type(8)))  float  v8f;

// ---- CDNA5 async global->LDS path (guarded: falls back if builtin absent) --
#if defined(__AMDGCN__) && __has_builtin(__builtin_amdgcn_global_load_async_to_lds_b128)
#define HAS_ASYNC_LDS 1
#else
#define HAS_ASYNC_LDS 0
#endif

#if HAS_ASYNC_LDS
// Builtin signature (from hipcc diagnostic): param0 is a pointer to a
// gcc-vector int4 in address space 1 (printed as "__device__" by HIP),
// param1 analogously in LDS (address space 3). Build exact-match typedefs
// and reach them via integer casts: generic->global is numerically
// identity; a generic LDS pointer carries the LDS offset in its low 32
// bits (flat aperture rule), and as(3) pointers are 32-bit wide.
typedef int v4i_gcc __attribute__((vector_size(16)));
typedef __attribute__((address_space(1))) v4i_gcc* as1_v4i;
typedef __attribute__((address_space(3))) v4i_gcc* as3_v4i;

__device__ __forceinline__ void async_ld16(const void* g, void* l) {
  __builtin_amdgcn_global_load_async_to_lds_b128(
      (as1_v4i)(uintptr_t)g,
      (as3_v4i)(unsigned int)(uintptr_t)l,
      0, 0);
}
__device__ __forceinline__ void wait_async0() {
#if __has_builtin(__builtin_amdgcn_s_wait_asynccnt)
  __builtin_amdgcn_s_wait_asynccnt(0);
#else
  asm volatile("s_wait_asynccnt 0" ::: "memory");
#endif
}
#endif

__device__ __forceinline__ v8bf ld8(const bf16_t* p) { return *(const v8bf*)p; }

__device__ __forceinline__ v16bf mk16(v8bf a, v8bf b) {
  v16bf r;
#pragma unroll
  for (int i = 0; i < 8; i++) { r[i] = a[i]; r[8 + i] = b[i]; }
  return r;
}

__device__ __forceinline__ v8f wmma_bf16(v16bf a, v16bf b, v8f c) {
  return __builtin_amdgcn_wmma_f32_16x16x32_bf16(
      /*neg_a=*/false, a, /*neg_b=*/false, b,
      /*c_mod=*/(short)0, c, /*reuse_a=*/false, /*reuse_b=*/false);
}

__device__ __forceinline__ float gelu_f(float x) {
  return 0.5f * x * (1.0f + erff(x * 0.70710678118654752f));
}

// =====================================================================
// Conv (grouped, K=128, pad 64/64, trim last) + GELU + residual
// =====================================================================
__global__ __launch_bounds__(256) void conv_gelu_res_kernel(
    const float* __restrict__ x, const float* __restrict__ cw,
    const float* __restrict__ cb, float* __restrict__ y)
{
  int idx = blockIdx.x * 256 + threadIdx.x;        // b*T*D + t*D + d
  int d = idx & (DIM - 1);
  int t = (idx / DIM) & (Tt - 1);
  int b = idx / (DIM * Tt);
  int g = d >> 6;                                   // group of 64 channels
  const float* wrow = cw + (size_t)d * 64 * KW;
  float acc = cb[d];
  int lo = (t >= 64) ? 0 : (64 - t);
  int hi = (t + 63 < Tt) ? KW : (Tt + 64 - t);
  for (int kk = lo; kk < hi; kk++) {
    const float* xr = x + ((size_t)(b * Tt + t + kk - 64)) * DIM + g * 64;
#pragma unroll 8
    for (int ci = 0; ci < 64; ci++)
      acc += xr[ci] * wrow[ci * KW + kk];
  }
  y[idx] = x[idx] + gelu_f(acc);
}

// =====================================================================
// Relative position bias table: posb[h][q][k]
// =====================================================================
__global__ __launch_bounds__(256) void relbias_kernel(
    const float* __restrict__ rel_e, float* __restrict__ posb)
{
  int idx = blockIdx.x * 256 + threadIdx.x;   // T*T
  int q = idx / Tt, k = idx % Tt;
  int rel = k - q;
  int ret = (rel > 0) ? 160 : 0;
  int ra = (rel < 0) ? -rel : rel;
  int ram = (ra < 1) ? 1 : ra;
  int large = 80 + (int)(logf((float)ram / 80.0f) / logf(10.0f) * 80.0f);
  if (large > 159) large = 159;
  ret += (ra < 80) ? ra : large;
#pragma unroll
  for (int hh = 0; hh < NH; hh++)
    posb[((size_t)hh * Tt + q) * Tt + k] = rel_e[ret * NH + hh];
}

// =====================================================================
// LayerNorm: one 256-thread block per row of D=1024
// =====================================================================
__global__ __launch_bounds__(256) void ln_kernel(
    const float* __restrict__ in, const float* __restrict__ w,
    const float* __restrict__ bias, float* __restrict__ out)
{
  __shared__ float red[256];
  const int row = blockIdx.x, tid = threadIdx.x;
  const float* p = in + (size_t)row * DIM;
  float s = 0.f;
  for (int i = tid; i < DIM; i += 256) s += p[i];
  red[tid] = s; __syncthreads();
  for (int st = 128; st > 0; st >>= 1) { if (tid < st) red[tid] += red[tid + st]; __syncthreads(); }
  float mean = red[0] * (1.0f / DIM);
  __syncthreads();
  float vs = 0.f;
  for (int i = tid; i < DIM; i += 256) { float d0 = p[i] - mean; vs += d0 * d0; }
  red[tid] = vs; __syncthreads();
  for (int st = 128; st > 0; st >>= 1) { if (tid < st) red[tid] += red[tid + st]; __syncthreads(); }
  float rs = rsqrtf(red[0] * (1.0f / DIM) + 1e-5f);
  for (int i = tid; i < DIM; i += 256)
    out[(size_t)row * DIM + i] = (p[i] - mean) * rs * w[i] + bias[i];
}

// =====================================================================
// Gate: one thread per (b,h,t)
// =====================================================================
__global__ __launch_bounds__(256) void gate_kernel(
    const float* __restrict__ hb, const float* __restrict__ gw,
    const float* __restrict__ gb, const float* __restrict__ gc,
    float* __restrict__ gateb)
{
  int idx = blockIdx.x * 256 + threadIdx.x;   // ((b*NH)+h)*Tt + t
  int t  = idx & (Tt - 1);
  int hh = (idx >> 9) & (NH - 1);
  int b  = idx >> 13;
  const float* g = hb + ((size_t)(b * Tt + t)) * DIM + hh * HDm;
  float s[8];
#pragma unroll
  for (int e = 0; e < 8; e++) {
    float a = gb[e];
    const float* wr = gw + e * HDm;
    for (int d = 0; d < HDm; d++) a += g[d] * wr[d];
    s[e] = a;
  }
  float pa = s[0] + s[1] + s[2] + s[3];
  float pb = s[4] + s[5] + s[6] + s[7];
  pa = 1.f / (1.f + __expf(-pa));
  pb = 1.f / (1.f + __expf(-pb));
  gateb[idx] = pa * (pb * gc[hh] - 1.f) + 2.f;
}

// =====================================================================
// WMMA GEMM: C[M,N] = act(A[M,K] @ W[N,K]^T + bias) (+ residual R)
// BM=BN=128, BK=32; fp32 tiles staged HBM->LDS with async DMA, then
// converted LDS->LDS to padded bf16 staging read as ISA WMMA fragments.
// =====================================================================
template<int ACT, int RES>
__global__ __launch_bounds__(256) void gemm_bias_kernel(
    const float* __restrict__ A, const float* __restrict__ W,
    const float* __restrict__ bias, const float* __restrict__ R,
    float* __restrict__ C, int Ndim, int Kdim)
{
  __shared__ float  Ag[128 * 32];     // fp32 async staging
  __shared__ float  Wg[128 * 32];
  __shared__ bf16_t As[128 * 48];     // padded bf16 WMMA staging
  __shared__ bf16_t Bs[128 * 48];
  const int tid = threadIdx.x;
  const int lane = tid & 31, wave = tid >> 5;
  const int wm = wave >> 2, wn = wave & 3;          // 2 x 4 wave grid
  const int m0 = blockIdx.y * 128, n0 = blockIdx.x * 128;
  const int lan = lane & 15;
  const int koffA = (lane < 16) ? 0 : 8;            // ISA A-frag K split
  const int koffB = (lane < 16) ? 0 : 16;           // unused; B frags use koffA
  v8f acc[4][2] = {};

  for (int k0 = 0; k0 < Kdim; k0 += 32) {
    __syncthreads();
#if HAS_ASYNC_LDS
    // 128x32 fp32 tile = 1024 x 16B chunks; 4 async chunks/thread/matrix
#pragma unroll
    for (int i = 0; i < 4; i++) {
      int c = i * 256 + tid;
      int r = c >> 3, col = (c & 7) * 4;
      async_ld16(&A[(size_t)(m0 + r) * Kdim + k0 + col], &Ag[r * 32 + col]);
      async_ld16(&W[(size_t)(n0 + r) * Kdim + k0 + col], &Wg[r * 32 + col]);
    }
    wait_async0();
#else
#pragma unroll
    for (int i = 0; i < 4; i++) {
      int c = i * 256 + tid;
      int r = c >> 3, col = (c & 7) * 4;
      *(float4*)&Ag[r * 32 + col] = *(const float4*)&A[(size_t)(m0 + r) * Kdim + k0 + col];
      *(float4*)&Wg[r * 32 + col] = *(const float4*)&W[(size_t)(n0 + r) * Kdim + k0 + col];
    }
#endif
    __syncthreads();
    if (k0 + 32 < Kdim) {                           // global_prefetch_b8
      int pr = tid >> 1, pc = (tid & 1) * 16;
      __builtin_prefetch(&A[(size_t)(m0 + pr) * Kdim + k0 + 32 + pc], 0, 1);
      __builtin_prefetch(&W[(size_t)(n0 + pr) * Kdim + k0 + 32 + pc], 0, 1);
    }
    // LDS->LDS fp32 -> bf16 into padded WMMA staging
#pragma unroll
    for (int i = 0; i < 16; i++) {
      int idx = i * 256 + tid;
      int r = idx >> 5, c = idx & 31;
      As[r * 48 + c] = (bf16_t)Ag[r * 32 + c];
      Bs[r * 48 + c] = (bf16_t)Wg[r * 32 + c];
    }
    __syncthreads();
    v16bf bfr[2];
#pragma unroll
    for (int j = 0; j < 2; j++) {
      const bf16_t* p = &Bs[(wn * 32 + j * 16 + lan) * 48 + koffA];
      bfr[j] = mk16(ld8(p), ld8(p + 16));
    }
#pragma unroll
    for (int i2 = 0; i2 < 4; i2++) {
      const bf16_t* p = &As[(wm * 64 + i2 * 16 + lan) * 48 + koffA];
      v16bf af = mk16(ld8(p), ld8(p + 16));
#pragma unroll
      for (int j = 0; j < 2; j++)
        acc[i2][j] = wmma_bf16(af, bfr[j], acc[i2][j]);
    }
  }
  const int grp8 = (lane >= 16) ? 8 : 0;
#pragma unroll
  for (int i2 = 0; i2 < 4; i2++)
#pragma unroll
    for (int j = 0; j < 2; j++)
#pragma unroll
      for (int r = 0; r < 8; r++) {
        int row = m0 + wm * 64 + i2 * 16 + r + grp8;
        int col = n0 + wn * 32 + j * 16 + lan;
        float v = acc[i2][j][r] + bias[col];
        if (ACT == 1) v = gelu_f(v);
        size_t o = (size_t)row * Ndim + col;
        if (RES) v += R[o];
        C[o] = v;
      }
}

// =====================================================================
// Fused flash-style attention. Block = (b, h, 128-row q tile); 8 waves,
// each owns 16 q rows. K/V tiles async-staged to LDS, S = QK^T (WMMA),
// + gate*pos_bias, online softmax, P round-trips per-wave LDS, O += P.V.
// =====================================================================
__global__ __launch_bounds__(256) void attn_kernel(
    const float* __restrict__ qv, const float* __restrict__ kv,
    const float* __restrict__ vv, const float* __restrict__ gateb,
    const float* __restrict__ posb, float* __restrict__ outb)
{
#if HAS_ASYNC_LDS
  __shared__ float Kf[128 * 64];      // fp32 async staging
  __shared__ float Vf[128 * 64];
#endif
  __shared__ bf16_t Ks[128 * 72];     // K tile, row-major [kr][d]
  __shared__ bf16_t Vt[64 * 136];     // V tile transposed [d][kr]
  __shared__ bf16_t Ps[8 * 16 * 136]; // per-wave P (16 x 128)
  const int tid = threadIdx.x;
  const int lane = tid & 31, wave = tid >> 5;
  const int lan = lane & 15;
  const int grp8 = (lane >= 16) ? 8 : 0;
  const int koffA = (lane < 16) ? 0 : 8;
  const int koffB = (lane < 16) ? 0 : 16;
  const int blk = blockIdx.x;
  const int qt = blk & 3, h = (blk >> 2) & 15, b = blk >> 6;
  const int q0 = qt * 128 + wave * 16;

  // Q fragments (scaled by 1/sqrt(HD)=0.125), held in regs all kernel
  v16bf qf[2];
  {
    const float* qp = qv + ((size_t)(b * Tt + q0 + lan)) * DIM + h * HDm;
#pragma unroll
    for (int kc = 0; kc < 2; kc++) {
      v16bf f;
#pragma unroll
      for (int j = 0; j < 8; j++) f[j]     = (bf16_t)(qp[kc * 32 + koffA + j]      * 0.125f);
#pragma unroll
      for (int j = 0; j < 8; j++) f[j + 8] = (bf16_t)(qp[kc * 32 + koffA + 16 + j] * 0.125f);
      qf[kc] = f;
    }
  }
  float gr[8], mrow[8], lrow[8];
#pragma unroll
  for (int r = 0; r < 8; r++) {
    gr[r] = gateb[((size_t)(b * NH + h)) * Tt + q0 + r + grp8];
    mrow[r] = -1e30f; lrow[r] = 0.f;
  }
  v8f oacc[4] = {};
  bf16_t* Pw = &Ps[wave * 16 * 136];

  for (int kt = 0; kt < 4; kt++) {
    __syncthreads();
#if HAS_ASYNC_LDS
    // 128x64 fp32 = 2048 x 16B chunks; 8 async chunks/thread/matrix
#pragma unroll
    for (int i = 0; i < 8; i++) {
      int c = i * 256 + tid;
      int kr = c >> 4, col = (c & 15) * 4;
      size_t g = ((size_t)(b * Tt + kt * 128 + kr)) * DIM + h * HDm + col;
      async_ld16(&kv[g], &Kf[kr * 64 + col]);
      async_ld16(&vv[g], &Vf[kr * 64 + col]);
    }
    wait_async0();
    __syncthreads();
#pragma unroll
    for (int i = 0; i < 32; i++) {     // cvt + transpose into WMMA staging
      int idx = i * 256 + tid;
      int kr = idx >> 6, d = idx & 63;
      Ks[kr * 72 + d]  = (bf16_t)Kf[kr * 64 + d];
      Vt[d * 136 + kr] = (bf16_t)Vf[kr * 64 + d];
    }
#else
#pragma unroll
    for (int i = 0; i < 32; i++) {     // stage 128x64 K and V^T (bf16)
      int idx = i * 256 + tid;
      int kr = idx >> 6, d = idx & 63;
      size_t g = ((size_t)(b * Tt + kt * 128 + kr)) * DIM + h * HDm + d;
      Ks[kr * 72 + d]  = (bf16_t)kv[g];
      Vt[d * 136 + kr] = (bf16_t)vv[g];
    }
#endif
    __syncthreads();

    // S tile (16 x 128) = Q . K^T
    v8f s[8];
#pragma unroll
    for (int nt = 0; nt < 8; nt++) {
      v8f a = {};
#pragma unroll
      for (int kc = 0; kc < 2; kc++) {
        const bf16_t* p = &Ks[(nt * 16 + lan) * 72 + kc * 32 + koffA];
        a = wmma_bf16(qf[kc], mk16(ld8(p), ld8(p + 16)), a);
      }
      s[nt] = a;
    }

    // bias + online softmax (half-wave shfl reductions over 16 lanes)
    float mloc[8];
#pragma unroll
    for (int r = 0; r < 8; r++) mloc[r] = -1e30f;
#pragma unroll
    for (int nt = 0; nt < 8; nt++)
#pragma unroll
      for (int r = 0; r < 8; r++) {
        int Rq = q0 + r + grp8;
        int Ck = kt * 128 + nt * 16 + lan;
        float val = s[nt][r] + gr[r] * posb[((size_t)h * Tt + Rq) * Tt + Ck];
        s[nt][r] = val;
        mloc[r] = fmaxf(mloc[r], val);
      }
#pragma unroll
    for (int off = 1; off < 16; off <<= 1)
#pragma unroll
      for (int r = 0; r < 8; r++) mloc[r] = fmaxf(mloc[r], __shfl_xor(mloc[r], off, 32));
    float alpha[8], lsum[8];
#pragma unroll
    for (int r = 0; r < 8; r++) {
      float mn = fmaxf(mrow[r], mloc[r]);
      alpha[r] = __expf(mrow[r] - mn);
      mrow[r] = mn;
      lsum[r] = 0.f;
    }
#pragma unroll
    for (int nt = 0; nt < 8; nt++)
#pragma unroll
      for (int r = 0; r < 8; r++) {
        float p = __expf(s[nt][r] - mrow[r]);
        s[nt][r] = p;
        lsum[r] += p;
      }
#pragma unroll
    for (int off = 1; off < 16; off <<= 1)
#pragma unroll
      for (int r = 0; r < 8; r++) lsum[r] += __shfl_xor(lsum[r], off, 32);
#pragma unroll
    for (int r = 0; r < 8; r++) lrow[r] = lrow[r] * alpha[r] + lsum[r];
#pragma unroll
    for (int ot = 0; ot < 4; ot++)
#pragma unroll
      for (int r = 0; r < 8; r++) oacc[ot][r] *= alpha[r];

    // P -> per-wave LDS (C-layout scatter), re-read as A-fragments
#pragma unroll
    for (int nt = 0; nt < 8; nt++)
#pragma unroll
      for (int r = 0; r < 8; r++)
        Pw[(r + grp8) * 136 + nt * 16 + lan] = (bf16_t)s[nt][r];

    // O += P (16x128) . V (128x64)
#pragma unroll
    for (int kc = 0; kc < 4; kc++) {
      const bf16_t* pp = &Pw[lan * 136 + kc * 32 + koffA];
      v16bf af = mk16(ld8(pp), ld8(pp + 16));
#pragma unroll
      for (int ot = 0; ot < 4; ot++) {
        const bf16_t* vp = &Vt[(ot * 16 + lan) * 136 + kc * 32 + koffA];
        oacc[ot] = wmma_bf16(af, mk16(ld8(vp), ld8(vp + 16)), oacc[ot]);
      }
    }
  }

#pragma unroll
  for (int ot = 0; ot < 4; ot++)
#pragma unroll
    for (int r = 0; r < 8; r++) {
      int Rq = q0 + r + grp8;
      int col = h * HDm + ot * 16 + lan;
      outb[((size_t)(b * Tt + Rq)) * DIM + col] = oacc[ot][r] / lrow[r];
    }
}

__global__ __launch_bounds__(256) void copy_kernel(
    const float* __restrict__ a, float* __restrict__ o, int n)
{
  int i = blockIdx.x * 256 + threadIdx.x;
  if (i < n) o[i] = a[i];
}

// =====================================================================
// Host orchestration
// =====================================================================
extern "C" void kernel_launch(void* const* d_in, const int* in_sizes, int n_in,
                              void* d_out, int out_size, void* d_ws, size_t ws_size,
                              hipStream_t stream)
{
  const float* x      = (const float*)d_in[0];
  const float* conv_w = (const float*)d_in[1];
  const float* conv_b = (const float*)d_in[2];
  const float* ln_w   = (const float*)d_in[3];
  const float* ln_b   = (const float*)d_in[4];
  const float* rel_e  = (const float*)d_in[5];
  const float* an_w   = (const float*)d_in[6];
  const float* an_b   = (const float*)d_in[7];
  const float* q_w    = (const float*)d_in[8];
  const float* q_b    = (const float*)d_in[9];
  const float* k_w    = (const float*)d_in[10];
  const float* k_b    = (const float*)d_in[11];
  const float* v_w    = (const float*)d_in[12];
  const float* v_b    = (const float*)d_in[13];
  const float* o_w    = (const float*)d_in[14];
  const float* o_b    = (const float*)d_in[15];
  const float* grl_w  = (const float*)d_in[16];
  const float* grl_b  = (const float*)d_in[17];
  const float* grc    = (const float*)d_in[18];
  const float* fn_w   = (const float*)d_in[19];
  const float* fn_b   = (const float*)d_in[20];
  const float* ffi_w  = (const float*)d_in[21];
  const float* ffi_b  = (const float*)d_in[22];
  const float* ffo_w  = (const float*)d_in[23];
  const float* ffo_b  = (const float*)d_in[24];

  float* ws = (float*)d_ws;
  const size_t MD_ = (size_t)Mrows * DIM;
  float* xb    = ws;                                 // residual stream
  float* hb    = xb + MD_;                           // LN output
  float* qb    = hb + MD_;
  float* kb    = qb + MD_;
  float* vb    = kb + MD_;
  float* ffb   = vb + MD_;                           // M x F
  float* posb  = ffb + (size_t)Mrows * FF;           // H x T x T
  float* gateb = posb + (size_t)NH * Tt * Tt;        // B x H x T
  float* ab    = hb;                                 // attn out aliases hb

  conv_gelu_res_kernel<<<(Bb * Tt * DIM) / 256, 256, 0, stream>>>(x, conv_w, conv_b, xb);
  relbias_kernel<<<(Tt * Tt) / 256, 256, 0, stream>>>(rel_e, posb);
  ln_kernel<<<Mrows, 256, 0, stream>>>(xb, ln_w, ln_b, xb);

  dim3 gD(DIM / 128, Mrows / 128);
  dim3 gF(FF / 128, Mrows / 128);
  for (int i = 0; i < LYR; i++) {
    ln_kernel<<<Mrows, 256, 0, stream>>>(xb, an_w + i * DIM, an_b + i * DIM, hb);
    gemm_bias_kernel<0,0><<<gD, 256, 0, stream>>>(hb, q_w + (size_t)i * DIM * DIM, q_b + i * DIM, nullptr, qb, DIM, DIM);
    gemm_bias_kernel<0,0><<<gD, 256, 0, stream>>>(hb, k_w + (size_t)i * DIM * DIM, k_b + i * DIM, nullptr, kb, DIM, DIM);
    gemm_bias_kernel<0,0><<<gD, 256, 0, stream>>>(hb, v_w + (size_t)i * DIM * DIM, v_b + i * DIM, nullptr, vb, DIM, DIM);
    gate_kernel<<<(Bb * NH * Tt) / 256, 256, 0, stream>>>(hb, grl_w + i * 8 * HDm, grl_b + i * 8, grc + i * NH, gateb);
    attn_kernel<<<Bb * NH * (Tt / 128), 256, 0, stream>>>(qb, kb, vb, gateb, posb, ab);
    gemm_bias_kernel<0,1><<<gD, 256, 0, stream>>>(ab, o_w + (size_t)i * DIM * DIM, o_b + i * DIM, xb, xb, DIM, DIM);
    ln_kernel<<<Mrows, 256, 0, stream>>>(xb, fn_w + i * DIM, fn_b + i * DIM, hb);
    gemm_bias_kernel<1,0><<<gF, 256, 0, stream>>>(hb, ffi_w + (size_t)i * FF * DIM, ffi_b + i * FF, nullptr, ffb, FF, DIM);
    gemm_bias_kernel<0,1><<<gD, 256, 0, stream>>>(ffb, ffo_w + (size_t)i * DIM * FF, ffo_b + i * DIM, xb, xb, DIM, FF);
  }
  copy_kernel<<<(out_size + 255) / 256, 256, 0, stream>>>(xb, (float*)d_out, out_size);
}
